// ObjectFusion_52415780880467
// MI455X (gfx1250) — matrix operand order repaired and checked by reading, not compiled, not run on hardware
//
#include <hip/hip_runtime.h>

// ---------------------------------------------------------------------------
// CDNA5 (gfx1250): bf16 WMMA GEMMs with async-to-LDS double buffering,
// VALU attention/LN glue. Shapes: B=8,F=64,M=20,N=512 frames,H=512,DFF=2048,
// HEADS=8 (d=64), 2 layers per encoder, two encoders (S=20 masked, S=2).
// ---------------------------------------------------------------------------

typedef __attribute__((ext_vector_type(16))) __bf16 v16bf;
typedef __attribute__((ext_vector_type(8)))  __bf16 v8bf;
typedef __attribute__((ext_vector_type(8)))  float  v8f;

#define NFRAMES 512
#define MOBJ    20
#define HDIM    512
#define DFF     2048
#define NEGV    -1e9f

// ---- CDNA5 async copy helpers (gfx1250: ASYNCcnt-tracked global->LDS DMA) ----
__device__ __forceinline__ void async_b128(unsigned lds_off, unsigned long long gaddr) {
  asm volatile("global_load_async_to_lds_b128 %0, %1, off"
               :: "v"(lds_off), "v"(gaddr) : "memory");
}
__device__ __forceinline__ void wait_async0() {
  asm volatile("s_wait_asynccnt 0x0" ::: "memory");
}

// ---------------------------------------------------------------------------
// Weight convert+transpose: W[K][Nc] f32 -> Wt[Nc][K] bf16 (K-contiguous rows).
// ---------------------------------------------------------------------------
__global__ void wt_cvt_kernel(const float* __restrict__ W, __bf16* __restrict__ Wt,
                              int K, int Nc) {
  int i = blockIdx.x * blockDim.x + threadIdx.x;
  if (i >= K * Nc) return;
  int k = i / Nc, n = i % Nc;
  Wt[(size_t)n * K + k] = (__bf16)W[i];
}

// ---------------------------------------------------------------------------
// Row LayerNorm f32 -> bf16 (generic D). One block (256 threads) per row.
// ---------------------------------------------------------------------------
__global__ void ln_rows_bf16_kernel(const float* __restrict__ X,
                                    const float* __restrict__ g,
                                    const float* __restrict__ b,
                                    __bf16* __restrict__ Y, int D) {
  int r = blockIdx.x, tid = threadIdx.x;
  const float* x = X + (size_t)r * D;
  __shared__ float s1[256], s2[256];
  float a = 0.f, q = 0.f;
  for (int d = tid; d < D; d += 256) { float t = x[d]; a += t; q += t * t; }
  s1[tid] = a; s2[tid] = q; __syncthreads();
  for (int w = 128; w > 0; w >>= 1) {
    if (tid < w) { s1[tid] += s1[tid + w]; s2[tid] += s2[tid + w]; }
    __syncthreads();
  }
  float mean = s1[0] / (float)D;
  float var  = s2[0] / (float)D - mean * mean;
  float inv  = rsqrtf(var + 1e-5f);
  __bf16* y = Y + (size_t)r * D;
  for (int d = tid; d < D; d += 256)
    y[d] = (__bf16)((x[d] - mean) * inv * g[d] + b[d]);
}

// ---------------------------------------------------------------------------
// Residual + LayerNorm (D=512): Out = LN(X + Yadd)*g + b, also bf16 copy.
// ---------------------------------------------------------------------------
__global__ void add_ln_kernel(const float* __restrict__ X,
                              const float* __restrict__ Yadd,
                              const float* __restrict__ g,
                              const float* __restrict__ b,
                              float* __restrict__ Out, __bf16* __restrict__ Obf) {
  int r = blockIdx.x, tid = threadIdx.x;   // 256 threads, 2 elems each
  const float* x = X + (size_t)r * HDIM;
  const float* y = Yadd + (size_t)r * HDIM;
  __shared__ float s1[256], s2[256];
  float t0 = x[tid] + y[tid];
  float t1 = x[tid + 256] + y[tid + 256];
  s1[tid] = t0 + t1; s2[tid] = t0 * t0 + t1 * t1; __syncthreads();
  for (int w = 128; w > 0; w >>= 1) {
    if (tid < w) { s1[tid] += s1[tid + w]; s2[tid] += s2[tid + w]; }
    __syncthreads();
  }
  float mean = s1[0] / (float)HDIM;
  float var  = s2[0] / (float)HDIM - mean * mean;
  float inv  = rsqrtf(var + 1e-5f);
  float o0 = (t0 - mean) * inv * g[tid] + b[tid];
  float o1 = (t1 - mean) * inv * g[tid + 256] + b[tid + 256];
  Out[(size_t)r * HDIM + tid]       = o0;
  Out[(size_t)r * HDIM + tid + 256] = o1;
  Obf[(size_t)r * HDIM + tid]       = (__bf16)o0;
  Obf[(size_t)r * HDIM + tid + 256] = (__bf16)o1;
}

// ---------------------------------------------------------------------------
// bf16 WMMA GEMM v3: C[M,N] = A[M,K] * Bt[N,K]^T + bias (opt ReLU, f32/bf16
// stores). Block 256 thr = 8 waves; block tile 128x64, K-stage 64, double-
// buffered LDS filled with GLOBAL_LOAD_ASYNC_TO_LDS_B128 (ASYNCcnt). Each
// wave owns a 16x64 strip: 4 accumulators share A fragments.
// v3: ALL fragments of a stage (2 A + 8 B = 20 ds_load_b128) are preloaded
// into registers before the 8-WMMA chain, so the DS wait happens once per
// stage and the WMMAs issue back-to-back into the matrix pipe.
// ---------------------------------------------------------------------------
#define BM 128
#define BN 64
#define BK 64
#define APITCH 72
#define BPITCH 72

__global__ void __launch_bounds__(256)
gemm_bf16_wmma(const __bf16* __restrict__ A, const __bf16* __restrict__ Bt,
               const float* __restrict__ bias, float* __restrict__ C,
               __bf16* __restrict__ Cbf, int M, int N, int K, int relu) {
  __shared__ __bf16 As[2][BM * APITCH];
  __shared__ __bf16 Bs[2][BN * BPITCH];

  int tid  = threadIdx.x;
  int lane = tid & 31;
  int wave = tid >> 5;
  int hi = lane >> 4, l15 = lane & 15;
  int tn0 = blockIdx.x * BN;
  int tm0 = blockIdx.y * BM;

  // Stage-load chunk assignment (16B chunks):
  // A stage: 128 rows x 32 bf16 = 512 chunks -> 2 per thread.
  // B stage:  64 rows x 32 bf16 = 256 chunks -> 1 per thread.
  int ac0 = 2 * tid, ac1 = 2 * tid + 1;
  int ar0 = ac0 >> 2, ak0 = (ac0 & 3) * 8;
  int ar1 = ac1 >> 2, ak1 = (ac1 & 3) * 8;
  int br  = tid >> 2, bk  = (tid & 3) * 8;

  auto load_stage = [&](int buf, int k0) {
    async_b128((unsigned)(uintptr_t)&As[buf][ar0 * APITCH + ak0],
               (unsigned long long)(uintptr_t)(A + (size_t)(tm0 + ar0) * K + k0 + ak0));
    async_b128((unsigned)(uintptr_t)&As[buf][ar1 * APITCH + ak1],
               (unsigned long long)(uintptr_t)(A + (size_t)(tm0 + ar1) * K + k0 + ak1));
    async_b128((unsigned)(uintptr_t)&Bs[buf][br * BPITCH + bk],
               (unsigned long long)(uintptr_t)(Bt + (size_t)(tn0 + br) * K + k0 + bk));
  };

  v8f acc[4] = {{}, {}, {}, {}};
  int wm = wave * 16;

  load_stage(0, 0);
  wait_async0();
  __syncthreads();

  union Frag { v16bf v; v8bf h[2]; };

  int nstage = K / BK;
  for (int s = 0; s < nstage; ++s) {
    int buf = s & 1;
    if (s + 1 < nstage) load_stage(buf ^ 1, (s + 1) * BK);   // overlap w/ compute

    // ---- preload all fragments of this stage (20 x ds_load_b128) ----
    Frag a[2], b[2][4];
#pragma unroll
    for (int t = 0; t < 2; ++t) {
      int kk = t * 32;
      const __bf16* arow = &As[buf][(wm + l15) * APITCH + kk];
      a[t].h[0] = *(const v8bf*)(arow + hi * 8);
      a[t].h[1] = *(const v8bf*)(arow + 16 + hi * 8);
#pragma unroll
      for (int nn = 0; nn < 4; ++nn) {
        const __bf16* brow = &Bs[buf][(nn * 16 + l15) * BPITCH + kk + hi * 8];
        b[t][nn].h[0] = *(const v8bf*)(brow);
        b[t][nn].h[1] = *(const v8bf*)(brow + 16);
      }
    }
    // ---- 8 independent WMMAs, back-to-back into the matrix pipe ----
#pragma unroll
    for (int t = 0; t < 2; ++t)
#pragma unroll
      for (int nn = 0; nn < 4; ++nn)
        acc[nn] = __builtin_amdgcn_wmma_f32_16x16x32_bf16(
            false, a[t].v, false, b[t][nn].v, (short)0, acc[nn], false, false);

    wait_async0();     // stage s+1 landed; safe to flip buffers
    __syncthreads();
  }

  int mb = tm0 + wm + hi * 8;
#pragma unroll
  for (int nn = 0; nn < 4; ++nn) {
    int n = tn0 + nn * 16 + l15;
    float bn = bias ? bias[n] : 0.f;
#pragma unroll
    for (int v = 0; v < 8; ++v) {
      float val = acc[nn][v] + bn;
      if (relu) val = fmaxf(val, 0.f);
      size_t o = (size_t)(mb + v) * N + n;
      if (C)   C[o]   = val;
      if (Cbf) Cbf[o] = (__bf16)val;
    }
  }
}

// ---------------------------------------------------------------------------
// Adaptive hard-attention mask. One block (128 thr) per frame.
// ---------------------------------------------------------------------------
__global__ void mask_kernel(const float* __restrict__ env,
                            const float* __restrict__ obj,
                            const int* __restrict__ omask,
                            float* __restrict__ padNeg, float* __restrict__ scale) {
  int n = blockIdx.x, tid = threadIdx.x;
  __shared__ float red[128];
  __shared__ float l2[MOBJ];
  const float* e = env + (size_t)n * HDIM;
  for (int m = 0; m < MOBJ; ++m) {
    const float* o = obj + ((size_t)n * MOBJ + m) * HDIM;
    float ss = 0.f;
    for (int d = tid; d < HDIM; d += 128) { float t = e[d] + o[d]; ss += t * t; }
    red[tid] = ss; __syncthreads();
    for (int w = 64; w > 0; w >>= 1) {
      if (tid < w) red[tid] += red[tid + w];
      __syncthreads();
    }
    if (tid == 0) l2[m] = sqrtf(red[0]);
    __syncthreads();
  }
  if (tid == 0) {
    bool mk[MOBJ]; float mcount = 0.f, mx = -1e30f;
    for (int m = 0; m < MOBJ; ++m) {
      mk[m] = omask[n * MOBJ + m] != 0;
      mcount += mk[m] ? 1.f : 0.f;
      float s = mk[m] ? l2[m] : NEGV;
      mx = fmaxf(mx, s);
    }
    float sm[MOBJ], den = 0.f;
    for (int m = 0; m < MOBJ; ++m) {
      sm[m] = __expf((mk[m] ? l2[m] : NEGV) - mx); den += sm[m];
    }
    float thresh = fminf(fmaxf(1.f / (1e-8f + mcount), 0.f), 1.f);
    bool hard[MOBJ]; float hcnt = 0.f;
    for (int m = 0; m < MOBJ; ++m) {
      float v = sm[m] / den * (mk[m] ? 1.f : 0.f);
      hard[m] = v >= thresh; hcnt += hard[m] ? 1.f : 0.f;
    }
    bool keep = hcnt > 0.f;
    for (int m = 0; m < MOBJ; ++m)
      padNeg[n * MOBJ + m] = (keep && !hard[m]) ? NEGV : 0.f;
    scale[n] = keep ? 1.f / fmaxf(hcnt, 1.f) : 0.f;
  }
}

// ---------------------------------------------------------------------------
// S=20 masked MHA (obj_fuser). One block per frame, 160 thr = 20 q x 8 heads.
// ---------------------------------------------------------------------------
__global__ void __launch_bounds__(160)
attn20_kernel(const float* __restrict__ Q, const float* __restrict__ K,
              const float* __restrict__ V, const float* __restrict__ padNeg,
              __bf16* __restrict__ Obf) {
  int n = blockIdx.x, tid = threadIdx.x;
  __shared__ float Ks[MOBJ * HDIM];
  const float* Kg = K + (size_t)n * MOBJ * HDIM;
  for (int i = tid; i < MOBJ * HDIM; i += 160) Ks[i] = Kg[i];
  __syncthreads();

  int h = tid / MOBJ, q = tid % MOBJ;
  const float* qp = Q + ((size_t)n * MOBJ + q) * HDIM + h * 64;
  float qr[64];
#pragma unroll
  for (int d = 0; d < 64; ++d) qr[d] = qp[d];

  float s[MOBJ]; float mx = -1e30f;
  for (int k = 0; k < MOBJ; ++k) {
    const float* kp = Ks + k * HDIM + h * 64;
    float acc = 0.f;
#pragma unroll
    for (int d = 0; d < 64; ++d) acc += qr[d] * kp[d];
    acc = acc * 0.125f + padNeg[n * MOBJ + k];
    s[k] = acc; mx = fmaxf(mx, acc);
  }
  float den = 0.f;
  for (int k = 0; k < MOBJ; ++k) { s[k] = __expf(s[k] - mx); den += s[k]; }
  float rden = 1.f / den;

  float o[64];
#pragma unroll
  for (int d = 0; d < 64; ++d) o[d] = 0.f;
  for (int k = 0; k < MOBJ; ++k) {
    const float* vp = V + ((size_t)n * MOBJ + k) * HDIM + h * 64;
    float a = s[k] * rden;
#pragma unroll
    for (int d = 0; d < 64; ++d) o[d] += a * vp[d];
  }
  __bf16* op = Obf + ((size_t)n * MOBJ + q) * HDIM + h * 64;
#pragma unroll
  for (int d = 0; d < 64; ++d) op[d] = (__bf16)o[d];
}

// ---------------------------------------------------------------------------
// S=2 MHA (eo_fuser). One thread per (frame, q, head).
// ---------------------------------------------------------------------------
__global__ void attn2_kernel(const float* __restrict__ Q, const float* __restrict__ K,
                             const float* __restrict__ V, __bf16* __restrict__ Obf,
                             int nframes) {
  int idx = blockIdx.x * blockDim.x + threadIdx.x;
  if (idx >= nframes * 16) return;
  int n = idx >> 4, r = idx & 15, q = r >> 3, h = r & 7;
  const float* qp = Q + ((size_t)n * 2 + q) * HDIM + h * 64;
  const float* k0 = K + ((size_t)n * 2 + 0) * HDIM + h * 64;
  const float* k1 = k0 + HDIM;
  float s0 = 0.f, s1 = 0.f;
#pragma unroll
  for (int d = 0; d < 64; ++d) { float qv = qp[d]; s0 += qv * k0[d]; s1 += qv * k1[d]; }
  s0 *= 0.125f; s1 *= 0.125f;
  float mx = fmaxf(s0, s1);
  float e0 = __expf(s0 - mx), e1 = __expf(s1 - mx);
  float rr = 1.f / (e0 + e1);
  float a0 = e0 * rr, a1 = e1 * rr;
  const float* v0 = V + ((size_t)n * 2 + 0) * HDIM + h * 64;
  const float* v1 = v0 + HDIM;
  __bf16* op = Obf + ((size_t)n * 2 + q) * HDIM + h * 64;
#pragma unroll
  for (int d = 0; d < 64; ++d) op[d] = (__bf16)(a0 * v0[d] + a1 * v1[d]);
}

// ---------------------------------------------------------------------------
// Pool kept objects + stack [env, fused] tokens (f32 + bf16).
// ---------------------------------------------------------------------------
__global__ void pool_stack_kernel(const float* __restrict__ Xobj,
                                  const float* __restrict__ env,
                                  const float* __restrict__ scale,
                                  float* __restrict__ S, __bf16* __restrict__ Sbf) {
  int n = blockIdx.x, tid = threadIdx.x;
  float sc = scale[n];
  for (int d = tid; d < HDIM; d += 256) {
    float acc = 0.f;
    for (int m = 0; m < MOBJ; ++m) acc += Xobj[((size_t)n * MOBJ + m) * HDIM + d];
    float fused = acc * sc;
    float e = env[(size_t)n * HDIM + d];
    size_t o0 = ((size_t)n * 2 + 0) * HDIM + d;
    size_t o1 = ((size_t)n * 2 + 1) * HDIM + d;
    S[o0] = e;      Sbf[o0] = (__bf16)e;
    S[o1] = fused;  Sbf[o1] = (__bf16)fused;
  }
}

__global__ void mean2_kernel(const float* __restrict__ S, float* __restrict__ out,
                             int total) {
  int i = blockIdx.x * blockDim.x + threadIdx.x;
  if (i >= total) return;
  int n = i >> 9, d = i & 511;
  out[i] = 0.5f * (S[((size_t)n * 2 + 0) * HDIM + d] +
                   S[((size_t)n * 2 + 1) * HDIM + d]);
}

// ---------------------------------------------------------------------------
// Host-side helpers
// ---------------------------------------------------------------------------
static void launch_gemm(const __bf16* A, const __bf16* Bt, const float* bias,
                        float* C, __bf16* Cbf, int M, int N, int K, int relu,
                        hipStream_t s) {
  dim3 grid(N / BN, M / BM), blk(256);
  gemm_bf16_wmma<<<grid, blk, 0, s>>>(A, Bt, bias, C, Cbf, M, N, K, relu);
}

struct FuserW {
  const float *Wq, *Wk, *Wv, *Wo, *bq, *bk, *bv, *bo, *W1, *b1, *W2, *b2;
  const float *ln1g, *ln1b, *ln2g, *ln2b;
};

static void run_encoder(float* X, __bf16* Xbf, int rows, const float* padNeg,
                        const FuserW& fw, __bf16* const wT[2][6],
                        float* Qf, float* Kf, float* Vf, __bf16* Obf,
                        float* Yf, __bf16* H1bf, hipStream_t s) {
  for (int li = 0; li < 2; ++li) {
    launch_gemm(Xbf, wT[li][0], fw.bq + li * HDIM, Qf, nullptr, rows, HDIM, HDIM, 0, s);
    launch_gemm(Xbf, wT[li][1], fw.bk + li * HDIM, Kf, nullptr, rows, HDIM, HDIM, 0, s);
    launch_gemm(Xbf, wT[li][2], fw.bv + li * HDIM, Vf, nullptr, rows, HDIM, HDIM, 0, s);
    if (padNeg) {
      attn20_kernel<<<rows / MOBJ, 160, 0, s>>>(Qf, Kf, Vf, padNeg, Obf);
    } else {
      int nframes = rows / 2;
      attn2_kernel<<<(nframes * 16 + 127) / 128, 128, 0, s>>>(Qf, Kf, Vf, Obf, nframes);
    }
    launch_gemm(Obf, wT[li][3], fw.bo + li * HDIM, Yf, nullptr, rows, HDIM, HDIM, 0, s);
    add_ln_kernel<<<rows, 256, 0, s>>>(X, Yf, fw.ln1g + li * HDIM, fw.ln1b + li * HDIM, X, Xbf);
    launch_gemm(Xbf, wT[li][4], fw.b1 + li * DFF, nullptr, H1bf, rows, DFF, HDIM, 1, s);
    launch_gemm(H1bf, wT[li][5], fw.b2 + li * HDIM, Yf, nullptr, rows, HDIM, DFF, 0, s);
    add_ln_kernel<<<rows, 256, 0, s>>>(X, Yf, fw.ln2g + li * HDIM, fw.ln2b + li * HDIM, X, Xbf);
  }
}

static void cvt_weight(const float* W, __bf16* Wt, int K, int Nc, hipStream_t s) {
  int tot = K * Nc;
  wt_cvt_kernel<<<(tot + 255) / 256, 256, 0, s>>>(W, Wt, K, Nc);
}

// ---------------------------------------------------------------------------
extern "C" void kernel_launch(void* const* d_in, const int* in_sizes, int n_in,
                              void* d_out, int out_size, void* d_ws, size_t ws_size,
                              hipStream_t stream) {
  (void)in_sizes; (void)n_in; (void)out_size; (void)ws_size;

  const float* env_feat = (const float*)d_in[0];
  const float* obj_feat = (const float*)d_in[1];
  const int*   obj_mask = (const int*)d_in[2];
  const float* env_ln_g = (const float*)d_in[3];
  const float* env_ln_b = (const float*)d_in[4];
  const float* obj_ln_g = (const float*)d_in[5];
  const float* obj_ln_b = (const float*)d_in[6];
  const float* W_env = (const float*)d_in[7];
  const float* b_env = (const float*)d_in[8];
  const float* W_obj = (const float*)d_in[9];
  const float* b_obj = (const float*)d_in[10];

  FuserW fu[2];
  for (int f = 0; f < 2; ++f) {
    int base = 11 + f * 16;
    fu[f].Wq  = (const float*)d_in[base + 0];  fu[f].Wk  = (const float*)d_in[base + 1];
    fu[f].Wv  = (const float*)d_in[base + 2];  fu[f].Wo  = (const float*)d_in[base + 3];
    fu[f].bq  = (const float*)d_in[base + 4];  fu[f].bk  = (const float*)d_in[base + 5];
    fu[f].bv  = (const float*)d_in[base + 6];  fu[f].bo  = (const float*)d_in[base + 7];
    fu[f].W1  = (const float*)d_in[base + 8];  fu[f].b1  = (const float*)d_in[base + 9];
    fu[f].W2  = (const float*)d_in[base + 10]; fu[f].b2  = (const float*)d_in[base + 11];
    fu[f].ln1g = (const float*)d_in[base + 12]; fu[f].ln1b = (const float*)d_in[base + 13];
    fu[f].ln2g = (const float*)d_in[base + 14]; fu[f].ln2b = (const float*)d_in[base + 15];
  }

  char* ws = (char*)d_ws;
  size_t off = 0;
  auto carve = [&](size_t bytes) -> char* {
    char* p = ws + off; off += (bytes + 255) & ~(size_t)255; return p;
  };
  const int NROWS = NFRAMES * MOBJ;   // 10240

  __bf16* WB_ENV = (__bf16*)carve((size_t)2048 * 512 * 2);
  __bf16* WB_OBJ = (__bf16*)carve((size_t)1024 * 512 * 2);
  __bf16* wT[2][2][6];
  for (int f = 0; f < 2; ++f)
    for (int li = 0; li < 2; ++li) {
      for (int w = 0; w < 4; ++w) wT[f][li][w] = (__bf16*)carve((size_t)512 * 512 * 2);
      wT[f][li][4] = (__bf16*)carve((size_t)512 * 2048 * 2);
      wT[f][li][5] = (__bf16*)carve((size_t)2048 * 512 * 2);
    }
  __bf16* A_ENV  = (__bf16*)carve((size_t)NFRAMES * 2048 * 2);
  __bf16* A_OBJ  = (__bf16*)carve((size_t)NROWS * 1024 * 2);
  float*  ENVP   = (float*)carve((size_t)NFRAMES * HDIM * 4);
  float*  X      = (float*)carve((size_t)NROWS * HDIM * 4);
  __bf16* Xbf    = (__bf16*)carve((size_t)NROWS * HDIM * 2);
  float*  Qf     = (float*)carve((size_t)NROWS * HDIM * 4);
  float*  Kf     = (float*)carve((size_t)NROWS * HDIM * 4);
  float*  Vf     = (float*)carve((size_t)NROWS * HDIM * 4);
  __bf16* Obf    = (__bf16*)carve((size_t)NROWS * HDIM * 2);
  float*  Yf     = (float*)carve((size_t)NROWS * HDIM * 4);
  __bf16* H1bf   = (__bf16*)carve((size_t)NROWS * DFF * 2);
  float*  PADNEG = (float*)carve((size_t)NFRAMES * MOBJ * 4);
  float*  SCALE  = (float*)carve((size_t)NFRAMES * 4);
  float*  STK    = (float*)carve((size_t)NFRAMES * 2 * HDIM * 4);
  __bf16* STKbf  = (__bf16*)carve((size_t)NFRAMES * 2 * HDIM * 2);

  // ---- 1) weights -> transposed bf16 ----
  cvt_weight(W_env, WB_ENV, 2048, 512, stream);
  cvt_weight(W_obj, WB_OBJ, 1024, 512, stream);
  for (int f = 0; f < 2; ++f)
    for (int li = 0; li < 2; ++li) {
      cvt_weight(fu[f].Wq + (size_t)li * 512 * 512,  wT[f][li][0], 512, 512, stream);
      cvt_weight(fu[f].Wk + (size_t)li * 512 * 512,  wT[f][li][1], 512, 512, stream);
      cvt_weight(fu[f].Wv + (size_t)li * 512 * 512,  wT[f][li][2], 512, 512, stream);
      cvt_weight(fu[f].Wo + (size_t)li * 512 * 512,  wT[f][li][3], 512, 512, stream);
      cvt_weight(fu[f].W1 + (size_t)li * 512 * 2048, wT[f][li][4], 512, 2048, stream);
      cvt_weight(fu[f].W2 + (size_t)li * 2048 * 512, wT[f][li][5], 2048, 512, stream);
    }

  // ---- 2) input LayerNorms -> bf16 activations ----
  ln_rows_bf16_kernel<<<NFRAMES, 256, 0, stream>>>(env_feat, env_ln_g, env_ln_b, A_ENV, 2048);
  ln_rows_bf16_kernel<<<NROWS,   256, 0, stream>>>(obj_feat, obj_ln_g, obj_ln_b, A_OBJ, 1024);

  // ---- 3) input projections (WMMA) ----
  launch_gemm(A_ENV, WB_ENV, b_env, ENVP, nullptr, NFRAMES, HDIM, 2048, 0, stream);
  launch_gemm(A_OBJ, WB_OBJ, b_obj, X, Xbf, NROWS, HDIM, 1024, 0, stream);

  // ---- 4) adaptive hard-attention mask ----
  mask_kernel<<<NFRAMES, 128, 0, stream>>>(ENVP, X, obj_mask, PADNEG, SCALE);

  // ---- 5) obj_fuser encoder (masked, S=20) ----
  run_encoder(X, Xbf, NROWS, PADNEG, fu[0], wT[0], Qf, Kf, Vf, Obf, Yf, H1bf, stream);

  // ---- 6) pool kept objects + stack [env, fused] ----
  pool_stack_kernel<<<NFRAMES, 256, 0, stream>>>(X, ENVP, SCALE, STK, STKbf);

  // ---- 7) eo_fuser encoder (S=2, no mask) ----
  run_encoder(STK, STKbf, NFRAMES * 2, nullptr, fu[1], wT[1], Qf, Kf, Vf, Obf, Yf, H1bf, stream);

  // ---- 8) mean over the two tokens -> output ----
  int total = NFRAMES * HDIM;
  mean2_kernel<<<(total + 255) / 256, 256, 0, stream>>>(STK, (float*)d_out, total);
}
